// ContextualLoss_54339926229668
// MI455X (gfx1250) — matrix operand order repaired and checked by reading, not compiled, not run on hardware
//
#include <hip/hip_runtime.h>
#include <hip/hip_bf16.h>
#include <math.h>

#define N_POS 4096
#define C_DIM 128
#define NB    2
#define EPS   1e-5f

typedef __attribute__((ext_vector_type(16))) __bf16 v16bf;
typedef __attribute__((ext_vector_type(8)))  float  v8f;

static __device__ __forceinline__ v8f wmma_bf16(v16bf a, v16bf b, v8f c) {
    // D = A(16x32 bf16) x B(32x16 bf16) + C(16x16 f32)
    return __builtin_amdgcn_wmma_f32_16x16x32_bf16(
        /*neg_a=*/false, a, /*neg_b=*/false, b,
        /*c_mod=*/(short)0, c, /*reuse_a=*/false, /*reuse_b=*/false);
}

// Load a 16-element bf16 fragment as two 16B chunks.
static __device__ __forceinline__ v16bf load_frag(const __bf16* p0, const __bf16* p1) {
    union U { v16bf v; uint4 q[2]; } u;
    u.q[0] = *reinterpret_cast<const uint4*>(p0);
    u.q[1] = *reinterpret_cast<const uint4*>(p1);
    return u.v;
}

// ---------- prep: [b,c,n] f32 -> [b,n,c] bf16 hi/lo split ----------
__global__ __launch_bounds__(256) void k_prep(const float* __restrict__ X,
                                              __bf16* __restrict__ Xh,
                                              __bf16* __restrict__ Xl) {
    unsigned i = blockIdx.x * 256u + threadIdx.x;          // exact: NB*C*N threads
    unsigned n = i & (N_POS - 1);
    unsigned bc = i >> 12;
    unsigned c = bc & (C_DIM - 1);
    unsigned b = bc >> 7;
    float x = X[i];
    __bf16 h = (__bf16)x;
    __bf16 l = (__bf16)(x - (float)h);
    size_t o = ((size_t)b * N_POS + n) * C_DIM + c;
    Xh[o] = h; Xl[o] = l;
}

// ---------- squared norms per position ----------
__global__ __launch_bounds__(256) void k_norms(const float* __restrict__ X,
                                               float* __restrict__ nrm) {
    unsigned i = blockIdx.x * 256u + threadIdx.x;          // exact: NB*N threads
    unsigned n = i & (N_POS - 1);
    unsigned b = i >> 12;
    const float* p = X + (size_t)b * C_DIM * N_POS + n;
    float s = 0.f;
    #pragma unroll 8
    for (int c = 0; c < C_DIM; ++c) { float v = p[(size_t)c * N_POS]; s += v * v; }
    nrm[i] = s;
}

// ---------- WMMA distance GEMM: D[t,s] = max(0, tsq[t] - 2 T.S^T + ssq[s]) ----------
__global__ __launch_bounds__(256) void k_gemm_dist(
    const __bf16* __restrict__ Th, const __bf16* __restrict__ Tl,
    const __bf16* __restrict__ Sh, const __bf16* __restrict__ Sl,
    const float*  __restrict__ tsq, const float* __restrict__ ssq,
    float* __restrict__ D)
{
    const int b    = blockIdx.z;
    const int lane = threadIdx.x & 31;
    const int wave = threadIdx.x >> 5;
    const int wm   = wave >> 1;           // 0..3 (M sub-tile)
    const int wn   = wave & 1;            // 0..1 (N sub-tile)
    const int m0   = blockIdx.y * 64 + wm * 16;
    const int n0   = blockIdx.x * 64 + wn * 32;
    const int l16  = lane & 15;
    const int hsel = lane >> 4;

    const size_t mat = (size_t)b * N_POS * C_DIM;
    // A frag: row m0+l16, K 0..15 (hsel=0) or 16..31 (hsel=1), contiguous
    const __bf16* arow_h = Th + mat + (size_t)(m0 + l16) * C_DIM + hsel * 16;
    const __bf16* arow_l = Tl + mat + (size_t)(m0 + l16) * C_DIM + hsel * 16;
    // B frag: column n (= S row n), K 0..15 (hsel=0) or 16..31 (hsel=1), contiguous
    const __bf16* b0_h = Sh + mat + (size_t)(n0 + l16) * C_DIM + hsel * 16;
    const __bf16* b0_l = Sl + mat + (size_t)(n0 + l16) * C_DIM + hsel * 16;
    const __bf16* b1_h = b0_h + 16 * C_DIM;
    const __bf16* b1_l = b0_l + 16 * C_DIM;

    v8f acc0 = {}; v8f acc1 = {};
    #pragma unroll
    for (int kc = 0; kc < C_DIM; kc += 32) {
        v16bf ah  = load_frag(arow_h + kc, arow_h + kc + 8);
        v16bf al  = load_frag(arow_l + kc, arow_l + kc + 8);
        v16bf bh0 = load_frag(b0_h + kc,  b0_h + kc + 8);
        v16bf bl0 = load_frag(b0_l + kc,  b0_l + kc + 8);
        v16bf bh1 = load_frag(b1_h + kc,  b1_h + kc + 8);
        v16bf bl1 = load_frag(b1_l + kc,  b1_l + kc + 8);
        // fp32-accurate product via hi/lo split: hh + hl + lh
        acc0 = wmma_bf16(ah, bh0, acc0);
        acc0 = wmma_bf16(ah, bl0, acc0);
        acc0 = wmma_bf16(al, bh0, acc0);
        acc1 = wmma_bf16(ah, bh1, acc1);
        acc1 = wmma_bf16(ah, bl1, acc1);
        acc1 = wmma_bf16(al, bh1, acc1);
    }

    // C/D layout: element r of v8f, lane L -> M = r + 8*(L>>4), N = L&15
    const int mrow = m0 + 8 * hsel;
    const float sq0 = ssq[(size_t)b * N_POS + n0 + l16];
    const float sq1 = ssq[(size_t)b * N_POS + n0 + 16 + l16];
    float* Db = D + (size_t)b * N_POS * N_POS;
    #pragma unroll
    for (int r = 0; r < 8; ++r) {
        float tq = tsq[(size_t)b * N_POS + mrow + r];
        float d0 = fmaxf(tq - 2.0f * acc0[r] + sq0, 0.0f);
        float d1 = fmaxf(tq - 2.0f * acc1[r] + sq1, 0.0f);
        Db[(size_t)(mrow + r) * N_POS + n0 + l16]      = d0;
        Db[(size_t)(mrow + r) * N_POS + n0 + 16 + l16] = d1;
    }
}

// ---------- block reduction (OP: 0=min, 1=max, 2=sum), result broadcast ----------
template<int OP>
static __device__ __forceinline__ float blockReduce(float v) {
    __shared__ float sm[8];
    #pragma unroll
    for (int o = 16; o > 0; o >>= 1) {
        float t = __shfl_xor(v, o, 32);
        v = (OP == 0) ? fminf(v, t) : (OP == 1) ? fmaxf(v, t) : (v + t);
    }
    if ((threadIdx.x & 31) == 0) sm[threadIdx.x >> 5] = v;
    __syncthreads();
    float r = sm[0];
    #pragma unroll
    for (int i = 1; i < 8; ++i)
        r = (OP == 0) ? fminf(r, sm[i]) : (OP == 1) ? fmaxf(r, sm[i]) : (r + sm[i]);
    __syncthreads();
    return r;
}

// ---------- per-row: m2[t] = rowmin, Z2[t] = rowsum of exp(2 - 2D/(m2+eps)) ----------
__global__ __launch_bounds__(256) void k_rowstats(const float* __restrict__ D,
                                                  float* __restrict__ m2,
                                                  float* __restrict__ Z2) {
    const int t = blockIdx.x, b = blockIdx.z;
    const float* row = D + ((size_t)b * N_POS + t) * N_POS;
    float vals[16];
    float mn = 3.0e38f;
    #pragma unroll
    for (int j = 0; j < 16; ++j) {
        vals[j] = row[j * 256 + threadIdx.x];
        mn = fminf(mn, vals[j]);
    }
    mn = blockReduce<0>(mn);
    const float inv = 2.0f / (mn + EPS);
    float z = 0.f;
    #pragma unroll
    for (int j = 0; j < 16; ++j) z += __expf(2.0f - vals[j] * inv);
    z = blockReduce<2>(z);
    if (threadIdx.x == 0) { m2[b * N_POS + t] = mn; Z2[b * N_POS + t] = z; }
}

// ---------- per-column strip: m1[s] = colmin, Z1[s] = colsum ----------
__global__ __launch_bounds__(256) void k_colstats(const float* __restrict__ D,
                                                  float* __restrict__ m1,
                                                  float* __restrict__ Z1) {
    const int s = blockIdx.x * 256 + threadIdx.x, b = blockIdx.z;
    const float* Db = D + (size_t)b * N_POS * N_POS;
    float mn = 3.0e38f;
    for (int t = 0; t < N_POS; ++t) mn = fminf(mn, Db[(size_t)t * N_POS + s]);
    const float inv = 2.0f / (mn + EPS);
    float z = 0.f;
    for (int t = 0; t < N_POS; ++t) z += __expf(2.0f - Db[(size_t)t * N_POS + s] * inv);
    m1[b * N_POS + s] = mn;
    Z1[b * N_POS + s] = z;
}

// ---------- dir1: r1[t] = rowmax(w1/Z1), accumulate mean into sim1[b] ----------
__global__ __launch_bounds__(256) void k_rowmax(const float* __restrict__ D,
                                                const float* __restrict__ m1,
                                                const float* __restrict__ Z1,
                                                float* __restrict__ sim1) {
    const int t = blockIdx.x, b = blockIdx.z;
    const float* row = D + ((size_t)b * N_POS + t) * N_POS;
    const float* m1b = m1 + b * N_POS;
    const float* z1b = Z1 + b * N_POS;
    float mx = 0.f;
    #pragma unroll
    for (int j = 0; j < 16; ++j) {
        int s = j * 256 + threadIdx.x;
        float w = __expf(2.0f - row[s] * (2.0f / (m1b[s] + EPS)));
        mx = fmaxf(mx, w / z1b[s]);
    }
    mx = blockReduce<1>(mx);
    if (threadIdx.x == 0) atomicAdd(&sim1[b], mx * (1.0f / N_POS));
}

// ---------- dir2: r2[s] = colmax(w2/Z2), accumulate mean into sim2[b] ----------
__global__ __launch_bounds__(256) void k_colmax(const float* __restrict__ D,
                                                const float* __restrict__ m2,
                                                const float* __restrict__ Z2,
                                                float* __restrict__ sim2) {
    const int s = blockIdx.x * 256 + threadIdx.x, b = blockIdx.z;
    const float* Db = D + (size_t)b * N_POS * N_POS;
    const float* m2b = m2 + b * N_POS;
    const float* z2b = Z2 + b * N_POS;
    float mx = 0.f;
    for (int t = 0; t < N_POS; ++t) {
        float w = __expf(2.0f - Db[(size_t)t * N_POS + s] * (2.0f / (m2b[t] + EPS)));
        mx = fmaxf(mx, w / z2b[t]);
    }
    float part = blockReduce<2>(mx);
    if (threadIdx.x == 0) atomicAdd(&sim2[b], part * (1.0f / N_POS));
}

__global__ void k_init(float* sims) {
    if (threadIdx.x < 4 && blockIdx.x == 0) sims[threadIdx.x] = 0.f;
}

__global__ void k_final(const float* __restrict__ sims, float* __restrict__ out) {
    if (threadIdx.x == 0 && blockIdx.x == 0) {
        // (mean_b(-log sim1) + mean_b(-log sim2)) / 2
        out[0] = 0.25f * (-logf(sims[0]) - logf(sims[1]) - logf(sims[2]) - logf(sims[3]));
    }
}

extern "C" void kernel_launch(void* const* d_in, const int* in_sizes, int n_in,
                              void* d_out, int out_size, void* d_ws, size_t ws_size,
                              hipStream_t stream) {
    (void)in_sizes; (void)n_in; (void)out_size; (void)ws_size;
    const float* src = (const float*)d_in[0];
    const float* tgt = (const float*)d_in[1];
    float* out = (float*)d_out;

    char* w = (char*)d_ws;
    auto carve = [&](size_t bytes) { void* p = (void*)w; w += (bytes + 255) & ~(size_t)255; return p; };
    float*  D    = (float*)carve((size_t)NB * N_POS * N_POS * sizeof(float)); // 128 MB
    __bf16* Th   = (__bf16*)carve((size_t)NB * N_POS * C_DIM * 2);
    __bf16* Tl   = (__bf16*)carve((size_t)NB * N_POS * C_DIM * 2);
    __bf16* Sh   = (__bf16*)carve((size_t)NB * N_POS * C_DIM * 2);
    __bf16* Sl   = (__bf16*)carve((size_t)NB * N_POS * C_DIM * 2);
    float*  tsq  = (float*)carve((size_t)NB * N_POS * 4);
    float*  ssq  = (float*)carve((size_t)NB * N_POS * 4);
    float*  m1   = (float*)carve((size_t)NB * N_POS * 4);
    float*  Z1   = (float*)carve((size_t)NB * N_POS * 4);
    float*  m2   = (float*)carve((size_t)NB * N_POS * 4);
    float*  Z2   = (float*)carve((size_t)NB * N_POS * 4);
    float*  sims = (float*)carve(4 * sizeof(float)); // sim1[2], sim2[2]

    k_init<<<1, 64, 0, stream>>>(sims);

    const int prepBlocks = (NB * C_DIM * N_POS) / 256;     // 4096
    k_prep<<<prepBlocks, 256, 0, stream>>>(tgt, Th, Tl);
    k_prep<<<prepBlocks, 256, 0, stream>>>(src, Sh, Sl);

    const int normBlocks = (NB * N_POS) / 256;             // 32
    k_norms<<<normBlocks, 256, 0, stream>>>(tgt, tsq);
    k_norms<<<normBlocks, 256, 0, stream>>>(src, ssq);

    k_gemm_dist<<<dim3(N_POS / 64, N_POS / 64, NB), 256, 0, stream>>>(
        Th, Tl, Sh, Sl, tsq, ssq, D);

    k_rowstats<<<dim3(N_POS, 1, NB), 256, 0, stream>>>(D, m2, Z2);
    k_colstats<<<dim3(N_POS / 256, 1, NB), 256, 0, stream>>>(D, m1, Z1);

    k_rowmax<<<dim3(N_POS, 1, NB), 256, 0, stream>>>(D, m1, Z1, sims);        // sim1 at sims[0..1]
    k_colmax<<<dim3(N_POS / 256, 1, NB), 256, 0, stream>>>(D, m2, Z2, sims + 2); // sim2 at sims[2..3]

    k_final<<<1, 32, 0, stream>>>(sims, out);
}
